// AFTLocalAttention_78391743086694
// MI455X (gfx1250) — compile-verified, matrix operand order checked by reading, not run on hardware
//
#include <hip/hip_runtime.h>
#include <hip/hip_bf16.h>

// ---------------------------------------------------------------------------
// AFT-local attention for MI455X (gfx1250), wave32 + v_wmma_f32_16x16x32_bf16.
// B=8, F=256, L=1024, H=4, D=256, MODEL=H*F=H*D=1024, WIN=128.
// ---------------------------------------------------------------------------

#define Bn   8
#define Fc   256
#define Ls   1024
#define Hn   4
#define Dh   256
#define MD   1024      // H*D == H*F
#define WINr 127       // window radius (WIN-1)

typedef __attribute__((ext_vector_type(16))) __bf16 v16bf;
typedef __attribute__((ext_vector_type(8)))  float  v8f;

static __device__ __forceinline__ v8f wmma_bf16(v16bf a, v16bf b, v8f c) {
    // 8 args: (neg_a, A, neg_b, B, c_mod, C, reuse_a, reuse_b)
    return __builtin_amdgcn_wmma_f32_16x16x32_bf16(false, a, false, b, (short)0, c, false, false);
}

// 16-bit A/B fragment map (ISA 7.12.2), wave32:
//   lane: row/col = lane & 15, hl = lane >> 4
//   elements 0..7  -> K = kc + hl*8 + e
//   elements 8..15 -> K = kc + 16 + hl*8 + (e-8)
static __device__ __forceinline__ v16bf ld_frag_bf16row(const __bf16* __restrict__ rowp,
                                                        int kc, int hl) {
    union U { uint4 u; __bf16 h[8]; } lo, hi;
    lo.u = *reinterpret_cast<const uint4*>(rowp + kc + hl * 8);
    hi.u = *reinterpret_cast<const uint4*>(rowp + kc + 16 + hl * 8);
    v16bf b;
#pragma unroll
    for (int e = 0; e < 8; ++e) { b[e] = lo.h[e]; b[8 + e] = hi.h[e]; }
    return b;
}

// ---------------------------------------------------------------------------
// P0: LDS-tiled repack of x [n,c,l,h] (f32) -> xpk [n,h,l,c] (bf16, c-contig).
// ---------------------------------------------------------------------------
__global__ void k_pack_x(const float* __restrict__ x, __bf16* __restrict__ xpk) {
    __shared__ __bf16 tile[32][33];
    int b = blockIdx.x;
    const int ct = b & 7;   b >>= 3;      // Fc/32
    const int lt = b & 31;  b >>= 5;      // Ls/32
    const int h  = b & 3;
    const int n  = b >> 2;
    const int l0 = lt * 32, c0 = ct * 32;
    const int t0 = threadIdx.x & 31;      // fast index
    const int tg = threadIdx.x >> 5;      // 0..7
#pragma unroll
    for (int j = 0; j < 4; ++j) {         // read: threads sweep l (16B stride)
        int c = tg + j * 8;
        tile[c][t0] = (__bf16)x[(((size_t)n * Fc + c0 + c) * Ls + l0 + t0) * Hn + h];
    }
    __syncthreads();
#pragma unroll
    for (int j = 0; j < 4; ++j) {         // write: threads sweep c (contiguous)
        int l = tg + j * 8;
        xpk[(((size_t)n * Hn + h) * Ls + l0 + l) * Fc + c0 + t0] = tile[t0][l];
    }
}

// ---------------------------------------------------------------------------
// P1: pack Wq|Wk|Wv [D,F] f32 -> bf16, concatenated [3][D][F].
// ---------------------------------------------------------------------------
__global__ void k_pack_w(const float* __restrict__ Wq, const float* __restrict__ Wk,
                         const float* __restrict__ Wv, __bf16* __restrict__ wpk) {
    int i = blockIdx.x * blockDim.x + threadIdx.x;       // 3*D*F
    int sel = i >> 16, r = i & 0xFFFF;                   // D*F = 65536
    const float* src = (sel == 0) ? Wq : (sel == 1) ? Wk : Wv;
    wpk[i] = (__bf16)src[r];
}

// ---------------------------------------------------------------------------
// P2: ew[t,s] = bf16( |t-s|<=127 ? exp(w_bias[t,s]) : 1.0 )   (mask -> exp(0)=1)
// ---------------------------------------------------------------------------
__global__ void k_prep_ew(const float* __restrict__ wb, __bf16* __restrict__ ew) {
    int i = blockIdx.x * blockDim.x + threadIdx.x;       // L*L
    int t = i >> 10, s = i & (Ls - 1);
    int d = t - s;
    bool in = (d <= WINr) && (d >= -WINr);
    ew[i] = (__bf16)(in ? __expf(wb[i]) : 1.0f);
}

// ---------------------------------------------------------------------------
// P3: out_wT[j,k] = bf16(out_w[k,j])
// ---------------------------------------------------------------------------
__global__ void k_prep_wT(const float* __restrict__ ow, __bf16* __restrict__ owT) {
    int i = blockIdx.x * blockDim.x + threadIdx.x;       // MD*MD
    int k = i >> 10, j = i & (MD - 1);
    owT[(size_t)j * MD + k] = (__bf16)ow[i];
}

// ---------------------------------------------------------------------------
// K1: fused QKV. One wave: 16(l) x 16(d) tile for one (n,h); one shared A
// fragment feeds THREE WMMAs per K-step (q,k,v).
//   sq   = sigmoid(q)                f32  [n, l, h*D+d]
//   ekT  = exp(k), ekvT = exp(k)*v   bf16 TRANSPOSED [n, h*D+d, l]
// ---------------------------------------------------------------------------
__global__ void k_qkv(const __bf16* __restrict__ xpk, const __bf16* __restrict__ wpk,
                      const float* __restrict__ Wqb, const float* __restrict__ Wkb,
                      const float* __restrict__ Wvb,
                      float* __restrict__ sq,
                      __bf16* __restrict__ ekT, __bf16* __restrict__ ekvT) {
    const int lane = threadIdx.x & 31;
    int w = blockIdx.x * (blockDim.x >> 5) + (threadIdx.x >> 5);
    const int dt = w & 15;  w >>= 4;            // Dh/16
    const int lt = w & 63;  w >>= 6;            // Ls/16
    const int h  = w & 3;
    const int n  = w >> 2;

    const int hl   = lane >> 4;
    const int row  = lt * 16 + (lane & 15);
    const int dcol = dt * 16 + (lane & 15);

    const __bf16* rowp = xpk + (((size_t)n * Hn + h) * Ls + row) * Fc;
    const __bf16* wq   = wpk + (size_t)dcol * Fc;
    const __bf16* wk   = wq + (size_t)Dh * Fc;
    const __bf16* wv   = wk + (size_t)Dh * Fc;

    v8f accq = {}, acck = {}, accv = {};
#pragma unroll
    for (int kc = 0; kc < Fc; kc += 32) {
        v16bf a  = ld_frag_bf16row(rowp, kc, hl);
        v16bf bq = ld_frag_bf16row(wq,   kc, hl);
        v16bf bk = ld_frag_bf16row(wk,   kc, hl);
        v16bf bv = ld_frag_bf16row(wv,   kc, hl);
        accq = wmma_bf16(a, bq, accq);
        acck = wmma_bf16(a, bk, acck);
        accv = wmma_bf16(a, bv, accv);
    }
    const float qb = Wqb[dcol], kb = Wkb[dcol], vb = Wvb[dcol];
    union U { uint4 u; __bf16 h[8]; } pk, pkv;
#pragma unroll
    for (int r = 0; r < 8; ++r) {
        int l = lt * 16 + hl * 8 + r;
        sq[((size_t)n * Ls + l) * MD + h * Dh + dcol] =
            1.0f / (1.0f + __expf(-(accq[r] + qb)));
        float ek = __expf(acck[r] + kb);
        pk.h[r]  = (__bf16)ek;
        pkv.h[r] = (__bf16)(ek * (accv[r] + vb));
    }
    size_t base = ((size_t)n * MD + h * Dh + dcol) * Ls + lt * 16 + hl * 8;
    *reinterpret_cast<uint4*>(ekT  + base) = pk.u;
    *reinterpret_cast<uint4*>(ekvT + base) = pkv.u;
}

// ---------------------------------------------------------------------------
// K2: per batch n: num = ew @ ekv, den = ew @ ek. One wave: 16(t) x 32(d) tile,
// shared A fragment feeds FOUR WMMAs per K-step. Fused y = sigmoid(q)*num/den.
// ---------------------------------------------------------------------------
__global__ void k_aft(const __bf16* __restrict__ ew,
                      const __bf16* __restrict__ ekT, const __bf16* __restrict__ ekvT,
                      const float* __restrict__ sq, __bf16* __restrict__ y) {
    const int lane = threadIdx.x & 31;
    int w = blockIdx.x * (blockDim.x >> 5) + (threadIdx.x >> 5);
    const int dt = w & 31;  w >>= 5;            // MD/32
    const int tt = w & 63;  w >>= 6;            // Ls/16
    const int n  = w;

    const int hl    = lane >> 4;
    const int trow  = tt * 16 + (lane & 15);
    const int dcol0 = dt * 32 + (lane & 15);
    const int dcol1 = dcol0 + 16;

    const __bf16* ewrow = ew   + (size_t)trow * Ls;
    const __bf16* ekp0  = ekT  + ((size_t)n * MD + dcol0) * Ls;
    const __bf16* ekp1  = ekT  + ((size_t)n * MD + dcol1) * Ls;
    const __bf16* ekvp0 = ekvT + ((size_t)n * MD + dcol0) * Ls;
    const __bf16* ekvp1 = ekvT + ((size_t)n * MD + dcol1) * Ls;

    v8f accN0 = {}, accN1 = {}, accD0 = {}, accD1 = {};
    for (int kc = 0; kc < Ls; kc += 32) {
        __builtin_prefetch(ewrow + kc + 256, 0, 0);     // global_prefetch_b8
        __builtin_prefetch(ekvp0 + kc + 256, 0, 0);
        __builtin_prefetch(ekp0  + kc + 256, 0, 0);
        v16bf a = ld_frag_bf16row(ewrow, kc, hl);
        accN0 = wmma_bf16(a, ld_frag_bf16row(ekvp0, kc, hl), accN0);
        accN1 = wmma_bf16(a, ld_frag_bf16row(ekvp1, kc, hl), accN1);
        accD0 = wmma_bf16(a, ld_frag_bf16row(ekp0,  kc, hl), accD0);
        accD1 = wmma_bf16(a, ld_frag_bf16row(ekp1,  kc, hl), accD1);
    }
#pragma unroll
    for (int r = 0; r < 8; ++r) {
        int tr = tt * 16 + hl * 8 + r;
        size_t i0 = ((size_t)n * Ls + tr) * MD + dcol0;
        y[i0]      = (__bf16)(sq[i0]      * accN0[r] / accD0[r]);
        y[i0 + 16] = (__bf16)(sq[i0 + 16] * accN1[r] / accD1[r]);
    }
}

// ---------------------------------------------------------------------------
// K3: h = y @ out_w + out_b + flatten(x). One wave: 16(l) x 32(j) tile.
// ---------------------------------------------------------------------------
__global__ void k_out(const __bf16* __restrict__ y, const __bf16* __restrict__ owT,
                      const float* __restrict__ ob, const float* __restrict__ x,
                      float* __restrict__ hbuf) {
    const int lane = threadIdx.x & 31;
    int w = blockIdx.x * (blockDim.x >> 5) + (threadIdx.x >> 5);
    const int jt = w & 31;  w >>= 5;            // MD/32
    const int lt = w & 63;  w >>= 6;            // Ls/16
    const int n  = w;

    const int hl    = lane >> 4;
    const int lrow  = lt * 16 + (lane & 15);
    const int jcol0 = jt * 32 + (lane & 15);
    const int jcol1 = jcol0 + 16;

    const __bf16* yp  = y   + ((size_t)n * Ls + lrow) * MD;
    const __bf16* wp0 = owT + (size_t)jcol0 * MD;
    const __bf16* wp1 = owT + (size_t)jcol1 * MD;

    v8f acc0 = {}, acc1 = {};
    for (int kc = 0; kc < MD; kc += 32) {
        __builtin_prefetch(yp + kc + 256, 0, 0);
        v16bf a = ld_frag_bf16row(yp, kc, hl);
        acc0 = wmma_bf16(a, ld_frag_bf16row(wp0, kc, hl), acc0);
        acc1 = wmma_bf16(a, ld_frag_bf16row(wp1, kc, hl), acc1);
    }
    // flatten (H outer, F inner): j = h*F + c
    const int hh0 = jcol0 >> 8, c0 = jcol0 & (Fc - 1);
    const int hh1 = jcol1 >> 8, c1 = jcol1 & (Fc - 1);
    const float b0 = ob[jcol0], b1 = ob[jcol1];
#pragma unroll
    for (int r = 0; r < 8; ++r) {
        int l = lt * 16 + hl * 8 + r;
        float r0 = x[(((size_t)n * Fc + c0) * Ls + l) * Hn + hh0];
        float r1 = x[(((size_t)n * Fc + c1) * Ls + l) * Hn + hh1];
        size_t o = ((size_t)n * Ls + l) * MD + jcol0;
        hbuf[o]      = acc0[r] + b0 + r0;
        hbuf[o + 16] = acc1[r] + b1 + r1;
    }
}

// ---------------------------------------------------------------------------
// K4: LayerNorm over last dim (1024). One 256-thread block per row.
// ---------------------------------------------------------------------------
__global__ void k_ln(const float* __restrict__ hbuf, const float* __restrict__ g,
                     const float* __restrict__ b, float* __restrict__ out) {
    __shared__ float s1[256], s2[256];
    const int row = blockIdx.x;
    const int tid = threadIdx.x;
    const float4 v = *reinterpret_cast<const float4*>(hbuf + (size_t)row * MD + tid * 4);
    s1[tid] = v.x + v.y + v.z + v.w;
    s2[tid] = v.x * v.x + v.y * v.y + v.z * v.z + v.w * v.w;
    __syncthreads();
#pragma unroll
    for (int off = 128; off > 0; off >>= 1) {
        if (tid < off) { s1[tid] += s1[tid + off]; s2[tid] += s2[tid + off]; }
        __syncthreads();
    }
    const float mu  = s1[0] * (1.0f / MD);
    const float var = s2[0] * (1.0f / MD) - mu * mu;
    const float rs  = rsqrtf(var + 1e-5f);
    float4 o;
    o.x = (v.x - mu) * rs * g[tid * 4 + 0] + b[tid * 4 + 0];
    o.y = (v.y - mu) * rs * g[tid * 4 + 1] + b[tid * 4 + 1];
    o.z = (v.z - mu) * rs * g[tid * 4 + 2] + b[tid * 4 + 2];
    o.w = (v.w - mu) * rs * g[tid * 4 + 3] + b[tid * 4 + 3];
    *reinterpret_cast<float4*>(out + (size_t)row * MD + tid * 4) = o;
}

// ---------------------------------------------------------------------------
extern "C" void kernel_launch(void* const* d_in, const int* in_sizes, int n_in,
                              void* d_out, int out_size, void* d_ws, size_t ws_size,
                              hipStream_t stream) {
    const float* x   = (const float*)d_in[0];
    const float* Wq  = (const float*)d_in[1];
    const float* Wqb = (const float*)d_in[2];
    const float* Wk  = (const float*)d_in[3];
    const float* Wkb = (const float*)d_in[4];
    const float* Wv  = (const float*)d_in[5];
    const float* Wvb = (const float*)d_in[6];
    const float* wb  = (const float*)d_in[7];
    const float* ow  = (const float*)d_in[8];
    const float* ob  = (const float*)d_in[9];
    const float* lng = (const float*)d_in[10];
    const float* lnb = (const float*)d_in[11];
    // d_in[12] (bool local_mask) intentionally unused: mask recomputed from indices.
    float* out = (float*)d_out;

    // workspace carve-out (~139 MB total)
    char* ws = (char*)d_ws;
    size_t off = 0;
    __bf16* xpk  = (__bf16*)(ws + off); off += (size_t)Bn * Hn * Ls * Fc * 2;  // packed x
    __bf16* wpk  = (__bf16*)(ws + off); off += (size_t)3 * Dh * Fc * 2;        // Wq|Wk|Wv bf16
    float*  sqb  = (float*)(ws + off);  off += (size_t)Bn * Ls * MD * 4;       // sigmoid(q)
    __bf16* ekT  = (__bf16*)(ws + off); off += (size_t)Bn * MD * Ls * 2;       // exp(k)   [n,col,l]
    __bf16* ekvT = (__bf16*)(ws + off); off += (size_t)Bn * MD * Ls * 2;       // exp(k)*v [n,col,l]
    __bf16* ewb  = (__bf16*)(ws + off); off += (size_t)Ls * Ls * 2;
    __bf16* ybf  = (__bf16*)(ws + off); off += (size_t)Bn * Ls * MD * 2;
    __bf16* owT  = (__bf16*)(ws + off); off += (size_t)MD * MD * 2;
    float*  hbuf = (float*)(ws + off);  off += (size_t)Bn * Ls * MD * 4;
    (void)ws_size; (void)in_sizes; (void)n_in; (void)out_size;

    k_pack_x<<<Bn * Hn * 32 * 8, 256, 0, stream>>>(x, xpk);
    k_pack_w<<<(3 * Dh * Fc) / 256, 256, 0, stream>>>(Wq, Wk, Wv, wpk);
    k_prep_ew<<<(Ls * Ls) / 256, 256, 0, stream>>>(wb, ewb);
    k_prep_wT<<<(MD * MD) / 256, 256, 0, stream>>>(ow, owT);

    // fused qkv: 8*4*64*16 = 32768 waves, 8 waves/block
    k_qkv<<<32768 / 8, 256, 0, stream>>>(xpk, wpk, Wqb, Wkb, Wvb, sqb, ekT, ekvT);

    // aft core: 8*64*32 = 16384 waves (16x32 tiles)
    k_aft<<<16384 / 8, 256, 0, stream>>>(ewb, ekT, ekvT, sqb, ybf);

    // out proj: 8*64*32 = 16384 waves (16x32 tiles)
    k_out<<<16384 / 8, 256, 0, stream>>>(ybf, owT, ob, x, hbuf);

    // layernorm: 8192 rows
    k_ln<<<Bn * Ls, 256, 0, stream>>>(hbuf, lng, lnb, out);
}
